// BayerFeatureExtractor_73924977098934
// MI455X (gfx1250) — matrix operand order repaired and checked by reading, compiled, not run on hardware
//
#include <hip/hip_runtime.h>

typedef __attribute__((ext_vector_type(16))) _Float16 v16h;
typedef __attribute__((ext_vector_type(8)))  float    v8f;

#define TW   64
#define THL  16
#define HALO 4
#define LW   (TW + 2*HALO)   // 72
#define LH   (THL + 2*HALO)  // 24

// 3x3 filter bank (rows 0..9): GREEN, GX, GY, GD1, GD2, LAP, HXX, HYY, HXY, BOX3
__device__ __constant__ float T3[10*9] = {
    0.f, 0.25f, 0.f,   0.25f, 0.f, 0.25f,   0.f, 0.25f, 0.f,          // GREEN
   -0.25f, 0.f, 0.25f, -0.5f, 0.f, 0.5f,   -0.25f, 0.f, 0.25f,        // GX
   -0.25f,-0.5f,-0.25f, 0.f, 0.f, 0.f,      0.25f, 0.5f, 0.25f,       // GY
    0.f, 0.f, 0.5f,    0.f, 0.f, 0.f,      -0.5f, 0.f, 0.f,           // GD1
    0.5f, 0.f, 0.f,    0.f, 0.f, 0.f,       0.f, 0.f,-0.5f,           // GD2
    0.f, 1.f, 0.f,     1.f,-4.f, 1.f,       0.f, 1.f, 0.f,            // LAP
    0.f, 0.f, 0.f,     1.f,-2.f, 1.f,       0.f, 0.f, 0.f,            // HXX
    0.f, 1.f, 0.f,     0.f,-2.f, 0.f,       0.f, 1.f, 0.f,            // HYY
    0.25f, 0.f,-0.25f, 0.f, 0.f, 0.f,      -0.25f, 0.f, 0.25f,        // HXY
    1.f/9.f,1.f/9.f,1.f/9.f,1.f/9.f,1.f/9.f,1.f/9.f,1.f/9.f,1.f/9.f,1.f/9.f
};
// 5-tap bank: HA2, HAI
__device__ __constant__ float T15[2*5] = {
    1.f, 0.f, -2.f, 0.f, 1.f,
   -0.25f, 0.5f, 0.5f, 0.5f, -0.25f
};

__device__ inline void normk(float* k, int n) {
    float m = 0.f;
    for (int i = 0; i < n; ++i) m += k[i];
    m /= (float)n;
    float s = 0.f;
    for (int i = 0; i < n; ++i) { k[i] -= m; s += fabsf(k[i]); }
    float inv = 1.f / fmaxf(s, 1e-6f);
    for (int i = 0; i < n; ++i) k[i] *= inv;
}

// Fill the 64-tap (7x7 padded to K=64) weight row for GEMM channel `row`.
// Channel map: 0..9 = 3x3 bank; 10 HA2H, 11 HAIH (1x5); 12 HA2V, 13 HAIV (5x1);
// 14 SINX, 15 SINY, 16 STRH, 17 STRV, 18 CHK, 19 DCT(2,2), 20 BOX5 (5x5);
// 21..23 Gabor theta = 0, pi/4, pi/2 (7x7). Rows >= 24: zero.
__device__ void fill_row(int row, float* w) {
    for (int i = 0; i < 64; ++i) w[i] = 0.f;
    if (row >= 24) return;
    const float PI = 3.14159265358979323846f;
    float k[49];
    if (row <= 9) {
        for (int a = 0; a < 3; ++a)
            for (int b = 0; b < 3; ++b)
                w[(a+2)*7 + (b+2)] = T3[row*9 + a*3 + b];
    } else if (row <= 11) {
        for (int b = 0; b < 5; ++b) w[3*7 + (b+1)] = T15[(row-10)*5 + b];
    } else if (row <= 13) {
        for (int a = 0; a < 5; ++a) w[(a+1)*7 + 3] = T15[(row-12)*5 + a];
    } else if (row <= 20) {
        for (int a = 0; a < 5; ++a)
            for (int b = 0; b < 5; ++b) {
                float v;
                if      (row == 14) v = sinf(2.f*PI*(float)b/5.f);
                else if (row == 15) v = sinf(2.f*PI*(float)a/5.f);
                else if (row == 16) v = (b % 2 == 0) ? 1.f : -1.f;
                else if (row == 17) v = (a % 2 == 0) ? 1.f : -1.f;
                else if (row == 18) v = ((a + b) % 2 == 0) ? 1.f : -1.f;
                else if (row == 19) v = cosf(PI*((float)a+0.5f)*2.f/5.f) *
                                        cosf(PI*((float)b+0.5f)*2.f/5.f);
                else                v = 1.f/25.f;   // BOX5 (no norm)
                k[a*5+b] = v;
            }
        if (row != 20) normk(k, 25);
        for (int a = 0; a < 5; ++a)
            for (int b = 0; b < 5; ++b)
                w[(a+1)*7 + (b+1)] = k[a*5+b];
    } else {
        float th = (float)(row - 21) * (PI * 0.25f);
        float ct = cosf(th), sn = sinf(th);
        for (int a = 0; a < 7; ++a)
            for (int b = 0; b < 7; ++b) {
                float ys = (float)(a - 3), xs = (float)(b - 3);
                float xt =  xs*ct + ys*sn;
                float yt = -xs*sn + ys*ct;
                float env = expf(-(xt*xt + 0.36f*yt*yt) / 5.12f);  // 2*sigma^2, gamma^2
                k[a*7+b] = env * cosf(2.f*PI*xt/3.5f);
            }
        normk(k, 49);
        for (int i = 0; i < 49; ++i) w[i] = k[i];
    }
}

__global__ __launch_bounds__(128)
void bayer_feats_kernel(const float* __restrict__ in, float* __restrict__ out,
                        int H, int W) {
    __shared__ float    sx[LH*LW];
    __shared__ _Float16 sh[LH*LW];
    __shared__ float    stg[4][24*16];

    const int tid  = threadIdx.x;
    const int wave = tid >> 5;
    const int lane = tid & 31;
    const int n    = lane & 15;   // N column (pixel within group)
    const int hl   = lane >> 4;   // lane half

    const int bx = blockIdx.x * TW;
    const int by = blockIdx.y * THL;
    const int bz = blockIdx.z;
    const size_t plane = (size_t)H * (size_t)W;
    const float* img = in + (size_t)bz * plane;

    // ---- load tile with reflect padding ----
    for (int i = tid; i < LH*LW; i += 128) {
        int r = i / LW, c = i - r*LW;
        int gy = by + r - HALO;
        int gx = bx + c - HALO;
        gy = gy < 0 ? -gy : gy;  gy = gy >= H ? 2*H - 2 - gy : gy;
        gx = gx < 0 ? -gx : gx;  gx = gx >= W ? 2*W - 2 - gx : gx;
        float v = img[(size_t)gy * W + gx];
        sx[i] = v;
        sh[i] = (_Float16)v;
    }

    // ---- build A matrices (weights) per ISA 16-bit A 16x32 layout ----
    // element e of lane-half hl -> in-chunk tap K' = 8*hl + e + (e>=8 ? 8 : 0)
    v16h a0[2], a1[2];
    {
        float wr[64];
        fill_row(n, wr);
        #pragma unroll
        for (int j = 0; j < 2; ++j)
            #pragma unroll
            for (int e = 0; e < 16; ++e) {
                int kk = 32*j + 8*hl + e + ((e < 8) ? 0 : 8);
                a0[j][e] = (_Float16)wr[kk];
            }
        fill_row(16 + n, wr);
        #pragma unroll
        for (int j = 0; j < 2; ++j)
            #pragma unroll
            for (int e = 0; e < 16; ++e) {
                int kk = 32*j + 8*hl + e + ((e < 8) ? 0 : 8);
                a1[j][e] = (_Float16)wr[kk];
            }
    }
    __syncthreads();

    float* st = &stg[wave][0];

    for (int pyy = 0; pyy < 4; ++pyy) {
        const int py = wave * 4 + pyy;
        for (int g = 0; g < 4; ++g) {
            const int px0 = g * 16;

            // ---- gather B (32x16 f16): lane-half hl holds K = 8*hl + e + (e>=8 ? 8 : 0) ----
            v16h b0, b1;
            #pragma unroll
            for (int e = 0; e < 16; ++e) {
                int k0 = 8*hl + e + ((e < 8) ? 0 : 8);            // < 32, always a valid tap
                int k1 = 32 + 8*hl + e + ((e < 8) ? 0 : 8);
                int ty0 = k0 / 7, tx0 = k0 - ty0*7;
                b0[e] = sh[(py + 1 + ty0)*LW + (px0 + n + 1 + tx0)];
                _Float16 v1 = (_Float16)0.f;
                if (k1 < 49) {
                    int ty1 = k1 / 7, tx1 = k1 - ty1*7;
                    v1 = sh[(py + 1 + ty1)*LW + (px0 + n + 1 + tx1)];
                }
                b1[e] = v1;
            }

            // ---- 4x WMMA: 32 channels x 16 pixels, K = 64 taps ----
            v8f c0 = {}, c1 = {};
            c0 = __builtin_amdgcn_wmma_f32_16x16x32_f16(false, a0[0], false, b0,
                                                        (short)0, c0, false, false);
            c0 = __builtin_amdgcn_wmma_f32_16x16x32_f16(false, a0[1], false, b1,
                                                        (short)0, c0, false, false);
            c1 = __builtin_amdgcn_wmma_f32_16x16x32_f16(false, a1[0], false, b0,
                                                        (short)0, c1, false, false);
            c1 = __builtin_amdgcn_wmma_f32_16x16x32_f16(false, a1[1], false, b1,
                                                        (short)0, c1, false, false);

            // ---- stage D to wave-private LDS: D row = r + 8*hl ----
            #pragma unroll
            for (int r = 0; r < 8; ++r) st[(8*hl + r)*16 + n] = c0[r];
            if (hl == 0) {
                #pragma unroll
                for (int r = 0; r < 8; ++r) st[(16 + r)*16 + n] = c1[r];
            }
            __syncthreads();

            // ---- per-pixel nonlinear post-processing (lanes 0..15) ----
            if (hl == 0) {
                const int px  = px0 + n;
                const int gxg = bx + px;
                const int gyg = by + py;
                const int xo  = (py + HALO)*LW + (px + HALO);

                float stv[24];
                #pragma unroll
                for (int c = 0; c < 24; ++c) stv[c] = st[c*16 + n];

                float x0 = sx[xo];
                const int ro = (gyg & 1);     // 1 = odd row
                const int co = (gxg & 1);     // 1 = odd col
                float green_est = (ro == co) ? x0 : stv[0];

                // rg/bg diff: denominators conv(mask,SMOOTH) == 0.25 exactly
                float accr = 0.f, accb = 0.f;
                const float sw1[3] = {0.25f, 0.5f, 0.25f};
                #pragma unroll
                for (int dy = -1; dy <= 1; ++dy) {
                    #pragma unroll
                    for (int dx = -1; dx <= 1; ++dx) {
                        int ry = (gyg + dy) & 1;
                        int cx = (gxg + dx) & 1;
                        float xv = sx[xo + dy*LW + dx];
                        float ge;
                        if (ry == cx) ge = xv;
                        else ge = 0.25f * (sx[xo + (dy-1)*LW + dx] + sx[xo + (dy+1)*LW + dx]
                                         + sx[xo + dy*LW + dx - 1] + sx[xo + dy*LW + dx + 1]);
                        float diff = xv - ge;
                        float wgt  = sw1[dy+1] * sw1[dx+1];
                        if (ry == 1 && cx == 0) accr += wgt * diff;   // r_mask
                        if (ry == 0 && cx == 1) accb += wgt * diff;   // b_mask
                    }
                }
                float rg_diff = 4.f * accr;
                float bg_diff = 4.f * accb;

                float gmag = sqrtf(stv[1]*stv[1] + stv[2]*stv[2] + 1e-8f);

                float vh = 0.f, vv = 0.f;
                #pragma unroll
                for (int d = -2; d <= 2; ++d) {
                    float mh = 0.2f*(sx[xo+d-2]+sx[xo+d-1]+sx[xo+d]+sx[xo+d+1]+sx[xo+d+2]);
                    float dh = sx[xo+d] - mh; vh += dh*dh;
                    float mv = 0.2f*(sx[xo+(d-2)*LW]+sx[xo+(d-1)*LW]+sx[xo+d*LW]
                                    +sx[xo+(d+1)*LW]+sx[xo+(d+2)*LW]);
                    float dv = sx[xo+d*LW] - mv; vv += dv*dv;
                }
                vh *= 0.2f; vv *= 0.2f;

                float res_h = x0 - stv[11];
                float res_v = x0 - stv[13];
                float sph   = sqrtf(stv[14]*stv[14] + stv[15]*stv[15] + 1e-8f);
                float gab   = fabsf(stv[21]) + fabsf(stv[22]) + fabsf(stv[23]);
                float rowp  = ro ? -1.f : 1.f;
                float colp  = co ? -1.f : 1.f;

                float o[30];
                o[0]=x0;       o[1]=green_est; o[2]=rg_diff;  o[3]=bg_diff;
                o[4]=stv[1];   o[5]=stv[2];    o[6]=stv[3];   o[7]=stv[4];
                o[8]=gmag;     o[9]=stv[5];    o[10]=stv[6];  o[11]=stv[7];
                o[12]=stv[8];  o[13]=vh;       o[14]=vv;      o[15]=stv[10];
                o[16]=stv[12]; o[17]=res_h;    o[18]=res_v;   o[19]=sph;
                o[20]=gab;     o[21]=stv[16];  o[22]=stv[17]; o[23]=stv[18];
                o[24]=stv[19]; o[25]=stv[9];   o[26]=stv[20];
                o[27]=rowp*x0; o[28]=colp*x0;  o[29]=rowp*colp*x0;

                float* op = out + (size_t)bz * 30 * plane + (size_t)gyg * W + gxg;
                #pragma unroll
                for (int c = 0; c < 30; ++c) op[(size_t)c * plane] = o[c];
            }
        }
    }
}

extern "C" void kernel_launch(void* const* d_in, const int* in_sizes, int n_in,
                              void* d_out, int out_size, void* d_ws, size_t ws_size,
                              hipStream_t stream) {
    (void)n_in; (void)out_size; (void)d_ws; (void)ws_size;
    const float* in = (const float*)d_in[0];
    float* out = (float*)d_out;
    const int H = 1024, W = 1024;
    const int NB = in_sizes[0] / (H * W);   // = 2
    dim3 grid(W / TW, H / THL, NB);
    bayer_feats_kernel<<<grid, 128, 0, stream>>>(in, out, H, W);
}
